// HierarchicalKANFFN_51934744543468
// MI455X (gfx1250) — matrix operand order repaired and checked by reading, not compiled, not run on hardware
//
#include <hip/hip_runtime.h>

#define D_MODEL 2048
#define NUM_TILES 64
#define D_SLICE 32
#define GRID_PTS 16
#define LN_EPS 1e-5f

typedef __attribute__((ext_vector_type(16))) __bf16 v16bf;
typedef __attribute__((ext_vector_type(8)))  float  v8f;

__device__ __forceinline__ float sgnf(float v) {
    return (v > 0.f) ? 1.f : ((v < 0.f) ? -1.f : 0.f);
}
// Order-preserving float<->uint encoding so atomicMin/atomicMax on uint give
// exact float min/max regardless of arrival order (deterministic).
__device__ __forceinline__ unsigned int encOrd(float f) {
    unsigned int u = __float_as_uint(f);
    return (u & 0x80000000u) ? ~u : (u | 0x80000000u);
}
__device__ __forceinline__ float decOrd(unsigned int k) {
    unsigned int u = (k & 0x80000000u) ? (k ^ 0x80000000u) : ~k;
    return __uint_as_float(u);
}

// ---------------------------------------------------------------------------
// Init: reset per-tile min/max keys; slope_sign[t*32+j] = sign(mean_g slopes)
// ---------------------------------------------------------------------------
__global__ __launch_bounds__(256) void hkan_init(
    const float* __restrict__ slopes, unsigned int* __restrict__ minK,
    unsigned int* __restrict__ maxK, float* __restrict__ sb)
{
    int tid = threadIdx.x;
    if (tid < NUM_TILES) { minK[tid] = 0xFFFFFFFFu; maxK[tid] = 0u; }
    for (int p = tid; p < NUM_TILES * D_SLICE; p += 256) {
        float s = 0.f;
        #pragma unroll
        for (int g = 0; g < GRID_PTS; ++g) s += slopes[p * GRID_PTS + g];
        sb[p] = sgnf(s * (1.0f / GRID_PTS));
    }
}

// ---------------------------------------------------------------------------
// Pass 1: 16 tokens per 256-thread block.
//   A) stream x: out=x copy + LN stats (per-wave, 2 tokens/wave)
//   B) per 512-dim chunk: bf16 signs -> LDS; wave0 runs 16 WMMAs to produce
//      tile scores for the chunk's 16 tiles (exact small-integer arithmetic)
//   C) argmax cluster then tile (first-max, matches jnp.argmax)
//   D) gather chosen 32 dims, fp32 xn, wave min/max, ordered atomics
// ---------------------------------------------------------------------------
__global__ __launch_bounds__(256) void hkan_pass1(
    const float* __restrict__ x, const float* __restrict__ gamma,
    const float* __restrict__ beta, const float* __restrict__ sb,
    float* __restrict__ out, unsigned int* __restrict__ minK,
    unsigned int* __restrict__ maxK, int* __restrict__ tileIdxG,
    float* __restrict__ xnSlice)
{
    __shared__ __bf16 sa[16][512];                 // 16 KB: chunk signs (A matrix)
    __shared__ float  sbLds[D_MODEL];              // 8 KB : slope signs
    __shared__ float  scoresLds[16][NUM_TILES];    // 4 KB
    __shared__ float  muLds[16], rsLds[16];
    __shared__ int    tIdxLds[16];

    const int tid  = threadIdx.x;
    const int lane = tid & 31;
    const int wave = tid >> 5;
    const long long blockTok = (long long)blockIdx.x * 16;

    for (int i = tid; i < D_MODEL; i += 256) sbLds[i] = sb[i];

    // ---- Stage A: copy + LN stats (wave w owns tokens 2w, 2w+1) ----
    for (int tt = 0; tt < 2; ++tt) {
        const int m = wave * 2 + tt;
        const long long n = blockTok + m;
        const float4* xr = (const float4*)(x + n * D_MODEL);
        float4* orow = (float4*)(out + n * D_MODEL);
        float s = 0.f, s2 = 0.f;
        #pragma unroll 4
        for (int i = lane; i < D_MODEL / 4; i += 32) {
            float4 v = xr[i];
            orow[i] = v;
            s  += v.x + v.y + v.z + v.w;
            s2 += v.x * v.x + v.y * v.y + v.z * v.z + v.w * v.w;
        }
        #pragma unroll
        for (int o = 16; o >= 1; o >>= 1) {
            s  += __shfl_xor(s,  o, 32);
            s2 += __shfl_xor(s2, o, 32);
        }
        if (lane == 0) {
            float mu  = s * (1.0f / D_MODEL);
            float var = s2 * (1.0f / D_MODEL) - mu * mu;
            muLds[m] = mu;
            rsLds[m] = rsqrtf(var + LN_EPS);
        }
    }
    __syncthreads();

    // ---- Stage B: 4 chunks of 512 dims; WMMA scoring ----
    for (int c = 0; c < 4; ++c) {
        for (int tt = 0; tt < 2; ++tt) {
            const int m = wave * 2 + tt;
            const long long n = blockTok + m;
            const float mu = muLds[m], rs = rsLds[m];
            const float4* xr = (const float4*)(x + n * D_MODEL + c * 512);
            const float4* g4 = (const float4*)(gamma + c * 512);
            const float4* b4 = (const float4*)(beta + c * 512);
            if (c < 3)  // warm next chunk (gfx1250 global_prefetch_b8)
                __builtin_prefetch(x + n * D_MODEL + (c + 1) * 512 + lane * 16, 0, 1);
            for (int i = lane; i < 128; i += 32) {
                float4 v = xr[i], g = g4[i], bb = b4[i];
                const int base = i * 4;
                float xn;
                xn = (v.x - mu) * rs * g.x + bb.x; sa[m][base + 0] = (__bf16)sgnf(xn);
                xn = (v.y - mu) * rs * g.y + bb.y; sa[m][base + 1] = (__bf16)sgnf(xn);
                xn = (v.z - mu) * rs * g.z + bb.z; sa[m][base + 2] = (__bf16)sgnf(xn);
                xn = (v.w - mu) * rs * g.w + bb.w; sa[m][base + 3] = (__bf16)sgnf(xn);
            }
        }
        __syncthreads();

        if (wave == 0) {  // whole wave active -> EXEC all ones for WMMA
            const int M    = lane & 15;   // A row / C row group
            const int half = lane >> 4;
            const int Ncol = lane & 15;   // B/C column
            v8f acc = {0.f, 0.f, 0.f, 0.f, 0.f, 0.f, 0.f, 0.f};
            for (int kk = 0; kk < 16; ++kk) {
                // A fragment: 16 bf16 per lane, layout per ISA 16-bit A 16x32
                v16bf a;
                const __bf16* p0 = &sa[M][kk * 32 + 8 * half];
                #pragma unroll
                for (int e = 0; e < 8; ++e) { a[e] = p0[e]; a[e + 8] = p0[16 + e]; }
                // B fragment: tile_sigs block-diagonal -> only column kk nonzero
                v16bf bf;
                if (Ncol == kk) {
                    const float* q = &sbLds[c * 512 + kk * 32 + 8 * half];
                    #pragma unroll
                    for (int e = 0; e < 8; ++e) { bf[e] = (__bf16)q[e]; bf[e + 8] = (__bf16)q[16 + e]; }
                } else {
                    #pragma unroll
                    for (int e = 0; e < 16; ++e) bf[e] = (__bf16)0.f;
                }
                acc = __builtin_amdgcn_wmma_f32_16x16x32_bf16(
                    false, a, false, bf, (short)0, acc, false, false);
            }
            // C/D layout: VGPR r -> row (r + 8*half), col = lane&15
            #pragma unroll
            for (int r = 0; r < 8; ++r)
                scoresLds[r + 8 * half][c * 16 + Ncol] = acc[r];
        }
        __syncthreads();
    }

    // ---- Stage C: argmax (one thread per token) ----
    if (tid < 16) {
        const int m = tid;
        float bestC = -1e30f; int bc = 0;
        for (int ci = 0; ci < 8; ++ci) {
            float s = 0.f;
            #pragma unroll
            for (int t = 0; t < 8; ++t) s += scoresLds[m][ci * 8 + t];
            if (s > bestC) { bestC = s; bc = ci; }
        }
        float bestT = -1e30f; int bt = bc * 8;
        for (int t = bc * 8; t < bc * 8 + 8; ++t) {
            float s = scoresLds[m][t];
            if (s > bestT) { bestT = s; bt = t; }
        }
        tIdxLds[m] = bt;
        tileIdxG[blockTok + m] = bt;
    }
    __syncthreads();

    // ---- Stage D: chosen-slice xn (fp32), min/max, publish ----
    for (int tt = 0; tt < 2; ++tt) {
        const int m = wave * 2 + tt;
        const long long n = blockTok + m;
        const int t = tIdxLds[m];
        const int d = t * 32 + lane;
        float v  = x[n * D_MODEL + d];
        float xn = (v - muLds[m]) * rsLds[m] * gamma[d] + beta[d];
        xnSlice[n * 32 + lane] = xn;
        float mn = xn, mx = xn;
        #pragma unroll
        for (int o = 16; o >= 1; o >>= 1) {
            mn = fminf(mn, __shfl_xor(mn, o, 32));
            mx = fmaxf(mx, __shfl_xor(mx, o, 32));
        }
        if (lane == 0) {
            atomicMin(&minK[t], encOrd(mn));
            atomicMax(&maxK[t], encOrd(mx));
        }
    }
}

// ---------------------------------------------------------------------------
// Pass 2: spline eval on the chosen 32 dims; out += delta. 8 tokens/block.
// ---------------------------------------------------------------------------
__global__ __launch_bounds__(256) void hkan_pass2(
    const float* __restrict__ bases, const float* __restrict__ slopes,
    const float* __restrict__ oscale, const unsigned int* __restrict__ minK,
    const unsigned int* __restrict__ maxK, const int* __restrict__ tileIdxG,
    const float* __restrict__ xnSlice, float* __restrict__ out, int N)
{
    const int tid = threadIdx.x;
    const int m = tid >> 5, j = tid & 31;
    const long long n = (long long)blockIdx.x * 8 + m;
    if (n >= N) return;
    const int t = tileIdxG[n];
    const float mn = decOrd(minK[t]);
    const float mx = decOrd(maxK[t]);
    const float xn = xnSlice[n * 32 + j];
    float u = (xn - mn) / (mx - mn + 1e-8f);
    u = fminf(fmaxf(u, 0.f), 1.f - 1e-6f);
    int gi = (int)(u * (float)GRID_PTS);
    gi = min(max(gi, 0), GRID_PTS - 1);
    const float xl = (u - (float)gi * (1.0f / GRID_PTS)) * (float)GRID_PTS;
    const int idx = t * (D_SLICE * GRID_PTS) + j * GRID_PTS + gi;
    const float delta = (bases[idx] + slopes[idx] * xl) * oscale[t];
    out[n * D_MODEL + t * 32 + j] += delta;
}

// ---------------------------------------------------------------------------
extern "C" void kernel_launch(void* const* d_in, const int* in_sizes, int n_in,
                              void* d_out, int out_size, void* d_ws, size_t ws_size,
                              hipStream_t stream)
{
    const float* x      = (const float*)d_in[0];
    const float* gamma  = (const float*)d_in[1];
    const float* beta   = (const float*)d_in[2];
    const float* bases  = (const float*)d_in[3];
    const float* slopes = (const float*)d_in[4];
    const float* oscale = (const float*)d_in[5];
    float* out = (float*)d_out;
    const int N = in_sizes[0] / D_MODEL;  // 16384

    char* ws = (char*)d_ws;
    unsigned int* minK = (unsigned int*)(ws);            // 64 * 4
    unsigned int* maxK = (unsigned int*)(ws + 256);      // 64 * 4
    float*        sb   = (float*)(ws + 512);             // 2048 * 4
    int*          tidx = (int*)(ws + 512 + 8192);        // N * 4
    float*        xns  = (float*)(ws + 512 + 8192 + (size_t)N * 4);  // N*32*4

    hkan_init <<<1, 256, 0, stream>>>(slopes, minK, maxK, sb);
    hkan_pass1<<<N / 16, 256, 0, stream>>>(x, gamma, beta, sb, out,
                                           minK, maxK, tidx, xns);
    hkan_pass2<<<(N + 7) / 8, 256, 0, stream>>>(bases, slopes, oscale,
                                                minK, maxK, tidx, xns, out, N);
}
